// HANLayer_51745765982967
// MI455X (gfx1250) — matrix operand order repaired and checked by reading, not compiled, not run on hardware
//
#include <hip/hip_runtime.h>
#include <math.h>
#include <stdint.h>

typedef __attribute__((ext_vector_type(16))) __bf16 v16bf;
typedef __attribute__((ext_vector_type(8)))  float  v8f;

struct FragU { uint4 lo, hi; };

__device__ __forceinline__ unsigned short f32_to_bf16(float f) {
  unsigned int u = __float_as_uint(f);
  u += 0x7FFFu + ((u >> 16) & 1u);   // round-to-nearest-even
  return (unsigned short)(u >> 16);
}

// ---------------- conversion / init kernels ----------------
__global__ void convert_f32_to_bf16(const float* __restrict__ x,
                                    unsigned short* __restrict__ y, int n) {
  int i = blockIdx.x * blockDim.x + threadIdx.x;
  if (i < n) y[i] = f32_to_bf16(x[i]);
}

// W [K,Ncol] row-major f32  ->  Wt [Ncol,K] row-major bf16
__global__ void transpose_f32_to_bf16(const float* __restrict__ W,
                                      unsigned short* __restrict__ Wt,
                                      int K, int Ncol) {
  int i = blockIdx.x * blockDim.x + threadIdx.x;
  if (i >= K * Ncol) return;
  int n = i % Ncol, k = i / Ncol;
  Wt[(size_t)n * K + k] = f32_to_bf16(W[i]);
}

__global__ void fill_f32(float* __restrict__ p, float v, int n) {
  int i = blockIdx.x * blockDim.x + threadIdx.x;
  if (i < n) p[i] = v;
}

// ---------------- WMMA GEMM: C[M,Ncol] = A[M,K] * Bt[Ncol,K]^T (+bias, relu) ----
// One wave computes a 16x64 tile of C (4 accumulators); A-fragment reused 4x.
template <int K, bool RELU, bool OUT_BF16>
__global__ __launch_bounds__(256) void gemm_bf16(
    const unsigned short* __restrict__ A,    // [M,K] bf16 bits
    const unsigned short* __restrict__ Bt,   // [Ncol,K] bf16 bits (B transposed)
    const float* __restrict__ bias,          // [Ncol] or nullptr
    float* __restrict__ Cf,                  // f32 out (if !OUT_BF16)
    unsigned short* __restrict__ Cb,         // bf16 out (if OUT_BF16)
    int M, int Ncol) {
  const int wave = threadIdx.x >> 5;
  const int lane = threadIdx.x & 31;
  const int row0 = (blockIdx.y * 8 + wave) * 16;
  if (row0 >= M) return;
  const int col0 = blockIdx.x * 64;
  const int lr = lane & 15;
  const int hi = lane >> 4;

  v8f acc[4] = {v8f{}, v8f{}, v8f{}, v8f{}};
  const unsigned short* Arow  = A  + (size_t)(row0 + lr) * K;
  const unsigned short* Bbase = Bt + (size_t)col0 * K;

#pragma unroll
  for (int k = 0; k < K; k += 32) {
    // A fragment (contiguous k order, same as B): lanes 0-15 -> K 0..15,
    // lanes 16-31 -> K 16..31
    FragU fa;
    const unsigned short* ap = Arow + k + hi * 16;
    fa.lo = *reinterpret_cast<const uint4*>(ap);
    fa.hi = *reinterpret_cast<const uint4*>(ap + 8);
    v16bf afrag = __builtin_bit_cast(v16bf, fa);
#pragma unroll
    for (int j = 0; j < 4; ++j) {
      // B fragment: col = lane%16 (col0+j*16), lanes 0-15 -> K k..k+15,
      // lanes 16-31 -> K k+16..k+31 (contiguous in transposed weight)
      const unsigned short* bp = Bbase + (size_t)(j * 16 + lr) * K + k + hi * 16;
      FragU fb;
      fb.lo = *reinterpret_cast<const uint4*>(bp);
      fb.hi = *reinterpret_cast<const uint4*>(bp + 8);
      v16bf bfrag = __builtin_bit_cast(v16bf, fb);
      acc[j] = __builtin_amdgcn_wmma_f32_16x16x32_bf16(
          false, afrag, false, bfrag, (short)0, acc[j], false, false);
    }
  }

  // D layout: VGPR r -> row row0 + r + 8*hi, col = col0 + j*16 + lane%16
#pragma unroll
  for (int j = 0; j < 4; ++j) {
    int col = col0 + j * 16 + lr;
    float bv = bias ? bias[col] : 0.0f;
#pragma unroll
    for (int r = 0; r < 8; ++r) {
      int row = row0 + r + hi * 8;
      float v = acc[j][r] + bv;
      if (RELU) v = v > 0.0f ? v : 0.0f;
      size_t o = (size_t)row * Ncol + col;
      if (OUT_BF16) Cb[o] = f32_to_bf16(v);
      else          Cf[o] = v;
    }
  }
}

// ---------------- attention logits: out[n,h] = sum_c h[n,h*64+c]*att[h,c] ----
__global__ void attn_dot(const float* __restrict__ h, const float* __restrict__ att,
                         float* __restrict__ out, int NH) {
  int i = blockIdx.x * blockDim.x + threadIdx.x;
  if (i >= NH) return;
  int hh = i & 3;
  int n  = i >> 2;
  const float4* hp = reinterpret_cast<const float4*>(h + (size_t)n * 256 + hh * 64);
  const float4* ap = reinterpret_cast<const float4*>(att + hh * 64);
  float s = 0.f;
#pragma unroll
  for (int t = 0; t < 16; ++t) {
    float4 a = hp[t], b = ap[t];
    s += a.x * b.x + a.y * b.y + a.z * b.z + a.w * b.w;
  }
  out[i] = s;
}

// ---------------- edge pass 1: leaky_relu logits + segment max ----------------
__global__ void edge_max(const int* __restrict__ src, const int* __restrict__ dst,
                         const float* __restrict__ a_s, const float* __restrict__ a_d,
                         float* __restrict__ ebuf, float* __restrict__ m, int EH) {
  int i = blockIdx.x * blockDim.x + threadIdx.x;
  if (i >= EH) return;
  int e = i >> 2, h = i & 3;
  int s = src[e], d = dst[e];
  float v = a_s[s * 4 + h] + a_d[d * 4 + h];
  v = v > 0.f ? v : 0.2f * v;           // leaky_relu(0.2)
  ebuf[i] = v;
  float* mp = m + d * 4 + h;            // sign-aware float atomic max
  if (v >= 0.f) atomicMax((int*)mp, __float_as_int(v));
  else          atomicMin((unsigned int*)mp, __float_as_uint(v));
}

// ---------------- edge pass 2: exp + segment sum -------------------------------
__global__ void edge_exp(const int* __restrict__ dst, const float* __restrict__ m,
                         float* __restrict__ ebuf, float* __restrict__ denom, int EH) {
  int i = blockIdx.x * blockDim.x + threadIdx.x;
  if (i >= EH) return;
  int e = i >> 2, h = i & 3;
  int d = dst[e];
  float ex = __expf(ebuf[i] - m[d * 4 + h]);
  ebuf[i] = ex;
  atomicAdd(denom + d * 4 + h, ex);
}

// ---------------- edge pass 3: msg = hs[src]*alpha, scatter-add ----------------
__global__ __launch_bounds__(256) void edge_agg(
    const int* __restrict__ src, const int* __restrict__ dst,
    const float* __restrict__ hs, const float* __restrict__ ex,
    const float* __restrict__ denom, float* __restrict__ agg, int E) {
  int t = blockIdx.x * blockDim.x + threadIdx.x;
  int e = t >> 6;                        // 64 threads per edge, float4 each
  if (e >= E) return;
  int j = t & 63;
  int h = j >> 4;                        // 16 float4 per head
  int s = src[e], d = dst[e];
  float alpha = ex[e * 4 + h] / denom[d * 4 + h];
  float4 v = reinterpret_cast<const float4*>(hs + (size_t)s * 256)[j];
  float* ap = agg + (size_t)d * 256 + j * 4;
  atomicAdd(ap + 0, v.x * alpha);
  atomicAdd(ap + 1, v.y * alpha);
  atomicAdd(ap + 2, v.z * alpha);
  atomicAdd(ap + 3, v.w * alpha);
}

// ---------------- finalize: head mean + bias, *etw, LayerNorm, ELU -------------
__global__ __launch_bounds__(256) void finalize(
    const float* __restrict__ agg, const float* __restrict__ bias,
    const float* __restrict__ lnw, const float* __restrict__ lnb,
    const float* __restrict__ etw, int etw_idx, float* __restrict__ out, int N) {
  int wave = threadIdx.x >> 5, lane = threadIdx.x & 31;
  int n = blockIdx.x * 8 + wave;
  if (n >= N) return;
  const float* a = agg + (size_t)n * 256;
  float w = etw[etw_idx];
  int c0 = lane, c1 = lane + 32;
  float v0 = 0.25f * (a[c0] + a[c0 + 64] + a[c0 + 128] + a[c0 + 192]) + bias[c0];
  float v1 = 0.25f * (a[c1] + a[c1 + 64] + a[c1 + 128] + a[c1 + 192]) + bias[c1];
  v0 *= w; v1 *= w;
  float s = v0 + v1;
#pragma unroll
  for (int o = 16; o > 0; o >>= 1) s += __shfl_xor(s, o, 32);
  float mu = s * (1.0f / 64.0f);
  float d0 = v0 - mu, d1 = v1 - mu;
  float q = d0 * d0 + d1 * d1;
#pragma unroll
  for (int o = 16; o > 0; o >>= 1) q += __shfl_xor(q, o, 32);
  float r = rsqrtf(q * (1.0f / 64.0f) + 1e-5f);
  float y0 = d0 * r * lnw[c0] + lnb[c0];
  float y1 = d1 * r * lnw[c1] + lnb[c1];
  y0 = y0 > 0.f ? y0 : __expf(y0) - 1.f;   // ELU
  y1 = y1 > 0.f ? y1 : __expf(y1) - 1.f;
  out[(size_t)n * 64 + c0] = y0;
  out[(size_t)n * 64 + c1] = y1;
}

// ---------------- launch -------------------------------------------------------
extern "C" void kernel_launch(void* const* d_in, const int* in_sizes, int n_in,
                              void* d_out, int out_size, void* d_ws, size_t ws_size,
                              hipStream_t stream) {
  const int N = 100000, E = 640000, IN = 128, HID = 256;

  const float* x_author = (const float*)d_in[0];
  const float* x_paper  = (const float*)d_in[1];
  const int*   ei_w     = (const int*)d_in[2];   // [2,E] writes: author->paper
  const int*   ei_wb    = (const int*)d_in[3];   // [2,E] written_by: paper->author
  const float* lwa  = (const float*)d_in[4];
  const float* lba  = (const float*)d_in[5];
  const float* lwp  = (const float*)d_in[6];
  const float* lbp  = (const float*)d_in[7];
  const float* g0ws = (const float*)d_in[8];
  const float* g0wd = (const float*)d_in[9];
  const float* g0as = (const float*)d_in[10];
  const float* g0ad = (const float*)d_in[11];
  const float* g0b  = (const float*)d_in[12];
  const float* g1ws = (const float*)d_in[13];
  const float* g1wd = (const float*)d_in[14];
  const float* g1as = (const float*)d_in[15];
  const float* g1ad = (const float*)d_in[16];
  const float* g1b  = (const float*)d_in[17];
  const float* lnwa = (const float*)d_in[18];
  const float* lnba = (const float*)d_in[19];
  const float* lnwp = (const float*)d_in[20];
  const float* lnbp = (const float*)d_in[21];
  const float* etw  = (const float*)d_in[22];

  char* p = (char*)d_ws;
  auto alloc = [&](size_t bytes) -> char* {
    char* r = p;
    p += (bytes + 255) & ~(size_t)255;
    return r;
  };

  unsigned short* xa16 = (unsigned short*)alloc((size_t)N * IN * 2);
  unsigned short* xp16 = (unsigned short*)alloc((size_t)N * IN * 2);
  unsigned short* wlat = (unsigned short*)alloc((size_t)IN * HID * 2);
  unsigned short* wlpt = (unsigned short*)alloc((size_t)IN * HID * 2);
  unsigned short* g0st = (unsigned short*)alloc((size_t)HID * HID * 2);
  unsigned short* g0dt = (unsigned short*)alloc((size_t)HID * HID * 2);
  unsigned short* g1st = (unsigned short*)alloc((size_t)HID * HID * 2);
  unsigned short* g1dt = (unsigned short*)alloc((size_t)HID * HID * 2);
  unsigned short* xa_h = (unsigned short*)alloc((size_t)N * HID * 2);  // relu(linear) author, bf16
  unsigned short* xp_h = (unsigned short*)alloc((size_t)N * HID * 2);
  float* hs0 = (float*)alloc((size_t)N * HID * 4);
  float* hs1 = (float*)alloc((size_t)N * HID * 4);
  float* hd0 = (float*)alloc((size_t)N * HID * 4);   // aliased as agg0 later
  float* hd1 = (float*)alloc((size_t)N * HID * 4);   // aliased as agg1 later
  float* a_s0 = (float*)alloc((size_t)N * 4 * 4);
  float* a_d0 = (float*)alloc((size_t)N * 4 * 4);
  float* a_s1 = (float*)alloc((size_t)N * 4 * 4);
  float* a_d1 = (float*)alloc((size_t)N * 4 * 4);
  float* m0   = (float*)alloc((size_t)N * 4 * 4);
  float* m1   = (float*)alloc((size_t)N * 4 * 4);
  float* den0 = (float*)alloc((size_t)N * 4 * 4);
  float* den1 = (float*)alloc((size_t)N * 4 * 4);
  float* e0   = (float*)alloc((size_t)E * 4 * 4);
  float* e1   = (float*)alloc((size_t)E * 4 * 4);
  float* agg0 = hd0;
  float* agg1 = hd1;

  auto g1 = [](int n) { return dim3((n + 255) / 256); };

  // 1) weights -> transposed bf16; inputs -> bf16
  transpose_f32_to_bf16<<<g1(IN * HID), 256, 0, stream>>>(lwa, wlat, IN, HID);
  transpose_f32_to_bf16<<<g1(IN * HID), 256, 0, stream>>>(lwp, wlpt, IN, HID);
  transpose_f32_to_bf16<<<g1(HID * HID), 256, 0, stream>>>(g0ws, g0st, HID, HID);
  transpose_f32_to_bf16<<<g1(HID * HID), 256, 0, stream>>>(g0wd, g0dt, HID, HID);
  transpose_f32_to_bf16<<<g1(HID * HID), 256, 0, stream>>>(g1ws, g1st, HID, HID);
  transpose_f32_to_bf16<<<g1(HID * HID), 256, 0, stream>>>(g1wd, g1dt, HID, HID);
  convert_f32_to_bf16<<<g1(N * IN), 256, 0, stream>>>(x_author, xa16, N * IN);
  convert_f32_to_bf16<<<g1(N * IN), 256, 0, stream>>>(x_paper,  xp16, N * IN);

  // 2) per-type linear + relu (WMMA, K=128)
  dim3 gg(HID / 64, (N / 16 + 7) / 8);
  gemm_bf16<128, true, true><<<gg, 256, 0, stream>>>(xa16, wlat, lba, nullptr, xa_h, N, HID);
  gemm_bf16<128, true, true><<<gg, 256, 0, stream>>>(xp16, wlpt, lbp, nullptr, xp_h, N, HID);

  // 3) GAT source/dest transforms (WMMA, K=256)
  gemm_bf16<256, false, false><<<gg, 256, 0, stream>>>(xa_h, g0st, nullptr, hs0, nullptr, N, HID);
  gemm_bf16<256, false, false><<<gg, 256, 0, stream>>>(xp_h, g0dt, nullptr, hd0, nullptr, N, HID);
  gemm_bf16<256, false, false><<<gg, 256, 0, stream>>>(xp_h, g1st, nullptr, hs1, nullptr, N, HID);
  gemm_bf16<256, false, false><<<gg, 256, 0, stream>>>(xa_h, g1dt, nullptr, hd1, nullptr, N, HID);

  // 4) attention logits per (node, head)
  attn_dot<<<g1(N * 4), 256, 0, stream>>>(hs0, g0as, a_s0, N * 4);
  attn_dot<<<g1(N * 4), 256, 0, stream>>>(hd0, g0ad, a_d0, N * 4);
  attn_dot<<<g1(N * 4), 256, 0, stream>>>(hs1, g1as, a_s1, N * 4);
  attn_dot<<<g1(N * 4), 256, 0, stream>>>(hd1, g1ad, a_d1, N * 4);

  // 5) init accumulators (every launch; hd0/hd1 now dead -> reuse as agg)
  fill_f32<<<g1(N * 4), 256, 0, stream>>>(m0, -INFINITY, N * 4);
  fill_f32<<<g1(N * 4), 256, 0, stream>>>(m1, -INFINITY, N * 4);
  fill_f32<<<g1(N * 4), 256, 0, stream>>>(den0, 0.f, N * 4);
  fill_f32<<<g1(N * 4), 256, 0, stream>>>(den1, 0.f, N * 4);
  fill_f32<<<g1(N * HID), 256, 0, stream>>>(agg0, 0.f, N * HID);
  fill_f32<<<g1(N * HID), 256, 0, stream>>>(agg1, 0.f, N * HID);

  // 6) edge type 0: author -writes-> paper
  const int* src0 = ei_w;       const int* dst0 = ei_w + E;
  const int* src1 = ei_wb;      const int* dst1 = ei_wb + E;
  edge_max<<<g1(E * 4), 256, 0, stream>>>(src0, dst0, a_s0, a_d0, e0, m0, E * 4);
  edge_exp<<<g1(E * 4), 256, 0, stream>>>(dst0, m0, e0, den0, E * 4);
  edge_agg<<<dim3((E * 64 + 255) / 256), 256, 0, stream>>>(src0, dst0, hs0, e0, den0, agg0, E);
  // edge type 1: paper -written_by-> author
  edge_max<<<g1(E * 4), 256, 0, stream>>>(src1, dst1, a_s1, a_d1, e1, m1, E * 4);
  edge_exp<<<g1(E * 4), 256, 0, stream>>>(dst1, m1, e1, den1, E * 4);
  edge_agg<<<dim3((E * 64 + 255) / 256), 256, 0, stream>>>(src1, dst1, hs1, e1, den1, agg1, E);

  // 7) finalize: d_out = [out_author (N*64) | out_paper (N*64)]
  float* out_author = (float*)d_out;
  float* out_paper  = (float*)d_out + (size_t)N * 64;
  finalize<<<dim3((N + 7) / 8), 256, 0, stream>>>(agg1, g1b, lnwa, lnba, etw, 1, out_author, N);
  finalize<<<dim3((N + 7) / 8), 256, 0, stream>>>(agg0, g0b, lnwp, lnbp, etw, 0, out_paper, N);
}
